// FDKN_28106265985611
// MI455X (gfx1250) — compile-verified, matrix operand order checked by reading, not run on hardware
//
#include <hip/hip_runtime.h>
#include <hip/hip_bf16.h>
#include <stdint.h>

// ---------------------------------------------------------------------------
// FDKN forward, CDNA5 (gfx1250) implementation.
// Convs as implicit-GEMM on v_wmma_f32_16x16x32_f16 (wave32):
//   M = B*H*W = 16384, N = Cout, K = Cin*ks*ks (padded to a multiple of 32)
// Wave tile: 32M x 32N (4 accumulators) when COUT%32==0, else 32M x 16N.
// Activations NHWC f16; weights BN-folded, packed [O][(ky*3+kx)*Cin+ci] f16
// with zero-padded K stride, so every 8-f16 K-chunk is a contiguous 16B
// global/ds/async transfer and the K-loop never needs a tail.
// Weight panels staged in LDS with GLOBAL_LOAD_ASYNC_TO_LDS_B128.
// ---------------------------------------------------------------------------

typedef _Float16 v8h  __attribute__((ext_vector_type(8)));
typedef _Float16 v16h __attribute__((ext_vector_type(16)));
typedef float    v8f  __attribute__((ext_vector_type(8)));

#define HW_PIX   16384      // 4 * 64 * 64
#define IMG_HW   65536      // 256*256

#if __has_builtin(__builtin_amdgcn_global_load_async_to_lds_b128) && \
    __has_builtin(__builtin_amdgcn_s_wait_asynccnt)
#define FDKN_ASYNC_LDS 1
#else
#define FDKN_ASYNC_LDS 0
#endif

#if FDKN_ASYNC_LDS
// Builtin signature (from clang diagnostic): both operands are pointers to
// 16-byte int vectors, global (AS1) source and LDS (AS3) destination.
typedef int fdkn_v4i __attribute__((vector_size(16)));
typedef __attribute__((address_space(1))) fdkn_v4i fdkn_gv4i;
typedef __attribute__((address_space(3))) fdkn_v4i fdkn_lv4i;
// generic->AS1 is an identity bit-pattern; generic LDS pointer's low 32 bits
// are the DS byte offset (flat aperture rules), so integer round-trips give
// legal address-space-qualified pointers without cross-AS pointer casts.
#define FDKN_GPTR(p) ((fdkn_gv4i*)(uintptr_t)(p))
#define FDKN_LPTR(p) ((fdkn_lv4i*)(uint32_t)(uintptr_t)(p))
#endif

// ---------------- workspace layout (bytes) ----------------
#define SZ_ACT      (16384u * 128u * 2u)          // 4,194,304
#define OFF_ACT_A   0u
#define OFF_ACT_B   (OFF_ACT_A + SZ_ACT)
#define OFF_ACT_IN  (OFF_ACT_B + SZ_ACT)          // 16384*16*2 = 524,288
#define OFF_IMK     (OFF_ACT_IN + 16384u*16u*2u)  // 16384*144*2 = 4,718,592
#define OFF_IMO     (OFF_IMK + 16384u*144u*2u)    // 16384*288*2 = 9,437,184
#define OFF_DPK     (OFF_IMO + 16384u*288u*2u)
#define OFF_DPO     (OFF_DPK + 16384u*144u*2u)
#define OFF_W       (OFF_DPO + 16384u*288u*2u)    // folded f16 weights, 2 towers
#define W_ELEMS_PER_TOWER 346112u                 // sum of per-layer O*KPAD
#define OFF_B       (OFF_W + 2u*W_ELEMS_PER_TOWER*2u)
#define B_ELEMS_PER_TOWER 880u
#define WS_NEED     (OFF_B + 2u*B_ELEMS_PER_TOWER*4u)

// ---------------------------------------------------------------------------
// BN-fold + repack with zero-padded K stride:
//   wOut[o*KPAD + (ky*ks+kx)*Cin + ci] = w[o][ci][ky][kx] * s[o]  (k < KT)
//   wOut[o*KPAD + k] = 0                                          (k >= KT)
//   bOut[o] = b[o]*s[o] + (be[o] - m[o]*s[o])   (s=1 if no BN)
// ---------------------------------------------------------------------------
__global__ void fdkn_fold_kernel(const float* __restrict__ w,
                                 const float* __restrict__ b,
                                 const float* __restrict__ g,
                                 const float* __restrict__ be,
                                 const float* __restrict__ m,
                                 const float* __restrict__ v,
                                 _Float16* __restrict__ wOut,
                                 float* __restrict__ bOut,
                                 int O, int CIN, int KS, int KPAD)
{
    int KT = CIN * KS * KS;
    int total = O * KPAD;
    int tid = blockIdx.x * blockDim.x + threadIdx.x;
    if (tid >= total) return;
    int o = tid / KPAD, k = tid - o * KPAD;
    float s = 1.f, sh = 0.f;
    if (g) { float inv = rsqrtf(v[o] + 1e-5f); s = g[o] * inv; sh = be[o] - m[o] * s; }
    float wv = 0.f;
    if (k < KT) {
        int tap = k / CIN, ci = k - tap * CIN;
        int ky = tap / KS, kx = tap - ky * KS;
        wv = w[((o * CIN + ci) * KS + ky) * KS + kx] * s;
    }
    wOut[(size_t)o * KPAD + k] = (_Float16)wv;
    if (k == 0) bOut[o] = b[o] * s + sh;
}

// ---------------------------------------------------------------------------
// gray + 4x4 pixel-unshuffle -> NHWC f16 (4,64,64,16); ch = i*4+j
// ---------------------------------------------------------------------------
__global__ void fdkn_resample_kernel(const float* __restrict__ src, int C,
                                     _Float16* __restrict__ dst)
{
    int tid = blockIdx.x * blockDim.x + threadIdx.x;
    if (tid >= 4 * 64 * 64 * 16) return;
    int ch = tid & 15;
    int x  = (tid >> 4) & 63;
    int y  = (tid >> 10) & 63;
    int bb = tid >> 16;
    int i = ch >> 2, j = ch & 3;
    int hh = y * 4 + i, ww = x * 4 + j;
    float val;
    if (C == 3) {
        size_t base = (size_t)bb * 3u * IMG_HW + (size_t)hh * 256 + ww;
        float c0 = src[base];
        float c1 = src[base + IMG_HW];
        float c2 = src[base + 2u * IMG_HW];
        val = 0.299f * c2 + 0.587f * c1 + 0.114f * c0;   // BGR-style coeffs
    } else {
        val = src[(size_t)bb * IMG_HW + (size_t)hh * 256 + ww];
    }
    dst[tid] = (_Float16)val;
}

// ---------------------------------------------------------------------------
// 8-f16 A-matrix chunk of the implicit im2col row (zero-padded 3x3, or 1x1).
// For kc in the zero-padded K tail (only layer 1) the gather may fetch a
// stale-but-in-bounds row; the matching B weights are zero, so it cancels.
// ---------------------------------------------------------------------------
template<int CIN, int KS>
__device__ __forceinline__ v8h fdkn_a_chunk(const _Float16* __restrict__ actIn,
                                            int m, int bb, int yy, int xx, int kc)
{
    if constexpr (KS == 1) {
        return *(const v8h*)(actIn + (size_t)m * CIN + kc);
    } else {
        const int tap = kc / CIN;
        const int cb  = kc - tap * CIN;       // cb % 8 == 0 -> 16B aligned
        const int ky  = tap / 3, kx = tap - ky * 3;
        const int iy  = yy + ky - 1, ix = xx + kx - 1;
        v8h vv = {};
        if ((unsigned)iy < 64u && (unsigned)ix < 64u)
            vv = *(const v8h*)(actIn +
                  (((size_t)bb * 4096u + (size_t)iy * 64u + ix) * CIN + cb));
        return vv;
    }
}

// ---------------------------------------------------------------------------
// Implicit-GEMM conv with WMMA f16 -> f32.
// Block = 128 threads (4 waves), waves stacked along M.
//   wave tile: 32M x (16*NTN)N; block tile: 128M x (16*NTN)N.
// Weight panel (16*NTN x KP f16) staged in LDS via async-to-LDS,
// shared by all 4 waves. ACT: 0 = relu, 1 = sigmoid, 2 = none.
// ---------------------------------------------------------------------------
template<int CIN, int COUT, int KS, int ACT>
__global__ __launch_bounds__(128)
void fdkn_conv_wmma(const _Float16* __restrict__ actIn,
                    const _Float16* __restrict__ wPack,
                    const float* __restrict__ bias,
                    _Float16* __restrict__ actOut)
{
    constexpr int KTOT = CIN * KS * KS;
    constexpr int KPAD = (KTOT + 31) & ~31;          // zero-padded K stride
    constexpr int NTN  = (COUT % 32 == 0) ? 2 : 1;   // N-tiles per wave
    constexpr int NB   = COUT / (16 * NTN);          // N-blocks in grid
    constexpr int NR   = 16 * NTN;                   // LDS weight rows
    constexpr int KP   = (NR * KPAD * 2 <= 49152) ? KPAD : (KPAD / 2);
    constexpr int NPANEL = KPAD / KP;
    __shared__ _Float16 ldsW[NR * KP];

    const int tid   = threadIdx.x;
    const int lane  = tid & 31;
    const int wave  = tid >> 5;
    const int nb    = blockIdx.x % NB;
    const int mb    = blockIdx.x / NB;
    const int mbase = (mb * 4 + wave) * 32;
    const int nbase = nb * NR;

    const int half = lane >> 4;     // K-half select per ISA 16-bit A/B layout
    const int lidx = lane & 15;

    // M decode for the wave's two 16-row tiles (NHWC (4,64,64,CIN))
    int mv[2], bbv[2], yyv[2], xxv[2];
    #pragma unroll
    for (int mt = 0; mt < 2; ++mt) {
        const int m = mbase + mt * 16 + lidx;
        mv[mt]  = m;
        bbv[mt] = m >> 12;
        yyv[mt] = (m >> 6) & 63;
        xxv[mt] = m & 63;
    }

    __builtin_prefetch(actIn + (size_t)mv[0] * CIN, 0, 1);

    v8f acc[2][NTN] = {};

    for (int pan = 0; pan < NPANEL; ++pan) {
        const int kp0 = pan * KP;
        if (pan) __syncthreads();          // protect LDS panel reuse

        // ---- stage 16*NTN x KP weight panel into LDS ----
        {
            constexpr int CH = (NR * KP) / 8;    // 16B chunks
            for (int c = tid; c < CH; c += 128) {
                const int n  = c / (KP / 8);
                const int k8 = c - n * (KP / 8);
                const _Float16* src =
                    wPack + (size_t)(nbase + n) * KPAD + kp0 + k8 * 8;
#if FDKN_ASYNC_LDS
                __builtin_amdgcn_global_load_async_to_lds_b128(
                    FDKN_GPTR(src), FDKN_LPTR(ldsW + c * 8), 0, 0);
#else
                ((uint4*)ldsW)[c] = *(const uint4*)src;
#endif
            }
#if FDKN_ASYNC_LDS
            __builtin_amdgcn_s_wait_asynccnt(0);
#endif
        }
        __syncthreads();

        #pragma unroll 2
        for (int k0 = 0; k0 < KP; k0 += 32) {
            const int kg = kp0 + k0;

            // ---- A fragments (implicit im2col gather from global) ----
            v16h A[2];
            #pragma unroll
            for (int mt = 0; mt < 2; ++mt) {
                v8h a0 = fdkn_a_chunk<CIN, KS>(actIn, mv[mt], bbv[mt], yyv[mt],
                                               xxv[mt], kg + 8 * half);
                v8h a1 = fdkn_a_chunk<CIN, KS>(actIn, mv[mt], bbv[mt], yyv[mt],
                                               xxv[mt], kg + 16 + 8 * half);
                #pragma unroll
                for (int i = 0; i < 8; ++i) { A[mt][i] = a0[i]; A[mt][8 + i] = a1[i]; }
            }

            // ---- B fragments from LDS (column n = lidx of each N-tile) ----
            v16h B[NTN];
            #pragma unroll
            for (int nt = 0; nt < NTN; ++nt) {
                const _Float16* wrow = ldsW + (nt * 16 + lidx) * KP;
                v8h b0 = *(const v8h*)(wrow + k0 + 8 * half);
                v8h b1 = *(const v8h*)(wrow + k0 + 16 + 8 * half);
                #pragma unroll
                for (int i = 0; i < 8; ++i) { B[nt][i] = b0[i]; B[nt][8 + i] = b1[i]; }
            }

            // ---- 4 (or 2) WMMAs reusing both A and B fragments ----
            #pragma unroll
            for (int mt = 0; mt < 2; ++mt)
                #pragma unroll
                for (int nt = 0; nt < NTN; ++nt)
                    acc[mt][nt] = __builtin_amdgcn_wmma_f32_16x16x32_f16(
                        false, A[mt], false, B[nt], (short)0, acc[mt][nt],
                        false, false);
        }
    }

    // ---- epilogue: bias + activation, scatter to NHWC f16 ----
    #pragma unroll
    for (int nt = 0; nt < NTN; ++nt) {
        const int n = nbase + nt * 16 + lidx;
        const float bv = bias[n];
        #pragma unroll
        for (int mt = 0; mt < 2; ++mt) {
            #pragma unroll
            for (int r = 0; r < 8; ++r) {
                const int mr = mbase + mt * 16 + r + 8 * half;  // C/D: row=vgpr+8*half
                float v = acc[mt][nt][r] + bv;
                if (ACT == 0)      v = v > 0.f ? v : 0.f;
                else if (ACT == 1) v = 1.f / (1.f + __expf(-v));
                actOut[(size_t)mr * COUT + n] = (_Float16)v;
            }
        }
    }
}

// ---------------------------------------------------------------------------
// Fused tail: head products -> pixel shuffle -> zero-mean kernel -> deformable
// bilinear sample of zero-padded 15x15 patch -> weighted sum -> residual.
// ---------------------------------------------------------------------------
__device__ __forceinline__ float fdkn_tap(const float* __restrict__ lrb,
                                          int h, int w, int xi, int yi)
{
    if ((unsigned)xi >= 15u || (unsigned)yi >= 15u) return 0.f;   // patch bounds
    const int ih = h - 7 + yi, iw = w - 7 + xi;                   // zero pad K//2
    if ((unsigned)ih >= 256u || (unsigned)iw >= 256u) return 0.f; // image bounds
    return lrb[ih * 256 + iw];
}

__global__ void fdkn_combine_kernel(const _Float16* __restrict__ imk,
                                    const _Float16* __restrict__ dpk,
                                    const _Float16* __restrict__ imo,
                                    const _Float16* __restrict__ dpo,
                                    const float* __restrict__ lr,
                                    float* __restrict__ out)
{
    int q = blockIdx.x * blockDim.x + threadIdx.x;
    if (q >= 4 * 256 * 256) return;
    const int w = q & 255, h = (q >> 8) & 255, b = q >> 16;
    const int Hq = h >> 2, i = h & 3, Wq = w >> 2, j = w & 3;
    const int p   = (b * 64 + Hq) * 64 + Wq;   // coarse pixel index
    const int sub = i * 4 + j;                 // pixel-shuffle sub-channel

    // 9 kernel weights = imk * dpk, zero-mean over taps
    float wt[9]; float mean = 0.f;
    #pragma unroll
    for (int c = 0; c < 9; ++c) {
        float v = (float)imk[(size_t)p * 144 + c * 16 + sub] *
                  (float)dpk[(size_t)p * 144 + c * 16 + sub];
        wt[c] = v; mean += v;
    }
    mean *= (1.f / 9.f);

    const float* lrb = lr + (size_t)b * IMG_HW;
    float accum = 0.f;
    #pragma unroll
    for (int a = 0; a < 3; ++a) {
        #pragma unroll
        for (int bb2 = 0; bb2 < 3; ++bb2) {
            const int c18 = (a * 3 + bb2) * 2;
            float o0 = (float)imo[(size_t)p * 288 + c18 * 16 + sub] *
                       (float)dpo[(size_t)p * 288 + c18 * 16 + sub];
            float o1 = (float)imo[(size_t)p * 288 + (c18 + 1) * 16 + sub] *
                       (float)dpo[(size_t)p * 288 + (c18 + 1) * 16 + sub];
            // grid_sample coords collapse to patch coords: lin[a]+off-0.5
            const float x = 7.f + (float)a   + o0 - 0.5f;
            const float y = 7.f + (float)bb2 + o1 - 0.5f;
            const float x0 = floorf(x), y0 = floorf(y);
            const float wx = x - x0, wy = y - y0;
            const int xi = (int)x0, yi = (int)y0;
            const float v00 = fdkn_tap(lrb, h, w, xi,     yi);
            const float v01 = fdkn_tap(lrb, h, w, xi + 1, yi);
            const float v10 = fdkn_tap(lrb, h, w, xi,     yi + 1);
            const float v11 = fdkn_tap(lrb, h, w, xi + 1, yi + 1);
            const float s = v00 * (1.f - wx) * (1.f - wy) + v01 * wx * (1.f - wy)
                          + v10 * (1.f - wx) * wy         + v11 * wx * wy;
            accum += s * (wt[a * 3 + bb2] - mean);
        }
    }
    out[q] = accum + lrb[h * 256 + w];
}

// ---------------------------------------------------------------------------
// Host-side orchestration
// ---------------------------------------------------------------------------
namespace {

struct LayerDesc { int wRel, bRel, bnRel; int O, CIN, KS; unsigned wOff, bOff; };

// relative indices inside one tower's 28 flat params; wOff is cumulative in
// f16 elements with K padded to a multiple of 32 (only layer 1: 144 -> 160).
static const LayerDesc kLayers[8] = {
    { 0,  1,  2,   32,  16, 3,      0,   0 },   // 32*160  = 5120
    { 6,  7, -1,   32,  32, 3,   5120,  32 },   // 32*288  = 9216
    { 8,  9, 10,   64,  32, 3,  14336,  64 },   // 64*288  = 18432
    {14, 15, -1,   64,  64, 3,  32768, 128 },   // 64*576  = 36864
    {16, 17, 18,  128,  64, 3,  69632, 192 },   // 128*576 = 73728
    {22, 23, -1,  128, 128, 3, 143360, 320 },   // 128*1152= 147456
    {24, 25, -1,  144, 128, 1, 290816, 448 },   // 144*128 = 18432
    {26, 27, -1,  288, 128, 1, 309248, 592 },   // 288*128 = 36864 -> total 346112
};

static void fold_tower(void* const* d_in, int base, _Float16* wreg, float* breg,
                       hipStream_t stream)
{
    for (int l = 0; l < 8; ++l) {
        const LayerDesc& L = kLayers[l];
        const float* w = (const float*)d_in[base + L.wRel];
        const float* b = (const float*)d_in[base + L.bRel];
        const float *g = nullptr, *be = nullptr, *m = nullptr, *v = nullptr;
        if (L.bnRel >= 0) {
            g  = (const float*)d_in[base + L.bnRel + 0];
            be = (const float*)d_in[base + L.bnRel + 1];
            m  = (const float*)d_in[base + L.bnRel + 2];
            v  = (const float*)d_in[base + L.bnRel + 3];
        }
        int kt   = L.CIN * L.KS * L.KS;
        int kpad = (kt + 31) & ~31;
        int total = L.O * kpad;
        int blocks = (total + 255) / 256;
        fdkn_fold_kernel<<<blocks, 256, 0, stream>>>(
            w, b, g, be, m, v, wreg + L.wOff, breg + L.bOff, L.O, L.CIN, L.KS, kpad);
    }
}

// block tile = 128 M rows x (COUT%32==0 ? 32 : 16) N cols
constexpr int conv_grid(int cout)
{
    return (16384 / 128) * ((cout % 32 == 0) ? cout / 32 : cout / 16);
}

static void run_tower(const _Float16* actIn, const _Float16* wreg, const float* breg,
                      _Float16* actA, _Float16* actB,
                      _Float16* headK, _Float16* headO, hipStream_t stream)
{
    fdkn_conv_wmma< 16,  32, 3, 0><<<conv_grid( 32), 128, 0, stream>>>(
        actIn, wreg + kLayers[0].wOff, breg + kLayers[0].bOff, actA);
    fdkn_conv_wmma< 32,  32, 3, 0><<<conv_grid( 32), 128, 0, stream>>>(
        actA,  wreg + kLayers[1].wOff, breg + kLayers[1].bOff, actB);
    fdkn_conv_wmma< 32,  64, 3, 0><<<conv_grid( 64), 128, 0, stream>>>(
        actB,  wreg + kLayers[2].wOff, breg + kLayers[2].bOff, actA);
    fdkn_conv_wmma< 64,  64, 3, 0><<<conv_grid( 64), 128, 0, stream>>>(
        actA,  wreg + kLayers[3].wOff, breg + kLayers[3].bOff, actB);
    fdkn_conv_wmma< 64, 128, 3, 0><<<conv_grid(128), 128, 0, stream>>>(
        actB,  wreg + kLayers[4].wOff, breg + kLayers[4].bOff, actA);
    fdkn_conv_wmma<128, 128, 3, 0><<<conv_grid(128), 128, 0, stream>>>(
        actA,  wreg + kLayers[5].wOff, breg + kLayers[5].bOff, actB);
    fdkn_conv_wmma<128, 144, 1, 1><<<conv_grid(144), 128, 0, stream>>>(
        actB,  wreg + kLayers[6].wOff, breg + kLayers[6].bOff, headK);
    fdkn_conv_wmma<128, 288, 1, 2><<<conv_grid(288), 128, 0, stream>>>(
        actB,  wreg + kLayers[7].wOff, breg + kLayers[7].bOff, headO);
}

} // namespace

extern "C" void kernel_launch(void* const* d_in, const int* in_sizes, int n_in,
                              void* d_out, int out_size, void* d_ws, size_t ws_size,
                              hipStream_t stream)
{
    (void)in_sizes; (void)n_in; (void)out_size;
    if (ws_size < (size_t)WS_NEED) return;   // scratch layout requires ~38.7 MB

    const float* image = (const float*)d_in[0];
    const float* lr    = (const float*)d_in[1];
    const int IM_BASE = 2, DP_BASE = 30;     // 28 flat params per tower

    char* ws = (char*)d_ws;
    _Float16* actA  = (_Float16*)(ws + OFF_ACT_A);
    _Float16* actB  = (_Float16*)(ws + OFF_ACT_B);
    _Float16* actIn = (_Float16*)(ws + OFF_ACT_IN);
    _Float16* imk   = (_Float16*)(ws + OFF_IMK);
    _Float16* imo   = (_Float16*)(ws + OFF_IMO);
    _Float16* dpk   = (_Float16*)(ws + OFF_DPK);
    _Float16* dpo   = (_Float16*)(ws + OFF_DPO);
    _Float16* wreg0 = (_Float16*)(ws + OFF_W);
    _Float16* wreg1 = wreg0 + W_ELEMS_PER_TOWER;
    float*    breg0 = (float*)(ws + OFF_B);
    float*    breg1 = breg0 + B_ELEMS_PER_TOWER;

    // BN-fold + f16 repack of both towers' weights
    fold_tower(d_in, IM_BASE, wreg0, breg0, stream);
    fold_tower(d_in, DP_BASE, wreg1, breg1, stream);

    const int RS_N = 4 * 64 * 64 * 16;

    // image tower
    fdkn_resample_kernel<<<(RS_N + 255) / 256, 256, 0, stream>>>(image, 3, actIn);
    run_tower(actIn, wreg0, breg0, actA, actB, imk, imo, stream);

    // depth tower
    fdkn_resample_kernel<<<(RS_N + 255) / 256, 256, 0, stream>>>(lr, 1, actIn);
    run_tower(actIn, wreg1, breg1, actA, actB, dpk, dpo, stream);

    // fused deformable-sample tail + residual
    const int NPIX = 4 * 256 * 256;
    fdkn_combine_kernel<<<(NPIX + 255) / 256, 256, 0, stream>>>(
        imk, dpk, imo, dpo, lr, (float*)d_out);
}
